// GraphSAGEModel_54443005444676
// MI455X (gfx1250) — matrix operand order repaired and checked by reading, not compiled, not run on hardware
//
#include <hip/hip_runtime.h>
#include <hip/hip_bf16.h>

// ---------------------------------------------------------------------------
// GraphSAGE (3 layers) for gfx1250: scatter-mean aggregation + bf16 WMMA GEMMs
// ---------------------------------------------------------------------------

typedef __attribute__((ext_vector_type(16))) __bf16 v16bf;
typedef __attribute__((ext_vector_type(8)))  float  v8f;

namespace {

constexpr int NNODES = 50000;
constexpr int NEDGES = 800000;
constexpr int KFEAT  = 128;   // IN_F == HID == 128 (uniform inner dim)

__device__ __forceinline__ unsigned short f32_to_bf16(float f) {
    unsigned int u = __float_as_uint(f);
    // round-to-nearest-even
    unsigned int r = u + 0x7FFFu + ((u >> 16) & 1u);
    return (unsigned short)(r >> 16);
}

union ABOp { v16bf v; uint4 q[2]; };

// ---------------- utility kernels ----------------

__global__ void k_zero_f32(float* __restrict__ p, int n) {
    int i = blockIdx.x * blockDim.x + threadIdx.x;
    if (i < n) p[i] = 0.0f;
}

__global__ void k_deg_count(const int* __restrict__ dst, float* __restrict__ deg, int e) {
    int i = blockIdx.x * blockDim.x + threadIdx.x;
    if (i < e) atomicAdd(deg + dst[i], 1.0f);
}

__global__ void k_invdeg(float* __restrict__ deg, int n) {
    int i = blockIdx.x * blockDim.x + threadIdx.x;
    if (i < n) deg[i] = 1.0f / fmaxf(deg[i], 1.0f);
}

// one wave (32 lanes) per edge; each lane handles 4 consecutive features
__global__ void k_scatter_add(const float* __restrict__ x,
                              const int* __restrict__ src,
                              const int* __restrict__ dst,
                              float* __restrict__ agg, int e) {
    int t    = blockIdx.x * blockDim.x + threadIdx.x;
    int edge = t >> 5;
    if (edge >= e) return;
    int lane = t & 31;
    int s = src[edge];
    int d = dst[edge];
    float4 v = *(const float4*)(x + (size_t)s * KFEAT + lane * 4);
    float* p = agg + (size_t)d * KFEAT + lane * 4;
    atomicAdd(p + 0, v.x);
    atomicAdd(p + 1, v.y);
    atomicAdd(p + 2, v.z);
    atomicAdd(p + 3, v.w);
}

__global__ void k_conv_bf16(const float* __restrict__ x,
                            unsigned short* __restrict__ o, int n) {
    int i = blockIdx.x * blockDim.x + threadIdx.x;
    if (i < n) o[i] = f32_to_bf16(x[i]);
}

// mean = agg * invdeg[node], converted to bf16
__global__ void k_conv_mean_bf16(const float* __restrict__ agg,
                                 const float* __restrict__ invdeg,
                                 unsigned short* __restrict__ o, int n) {
    int i = blockIdx.x * blockDim.x + threadIdx.x;
    if (i < n) o[i] = f32_to_bf16(agg[i] * invdeg[i >> 7]);  // KFEAT == 128
}

// transpose-convert weights [K][C] (row-major) -> bf16 [C][K]
__global__ void k_conv_wt(const float* __restrict__ w,
                          unsigned short* __restrict__ wt, int k, int c) {
    int i = blockIdx.x * blockDim.x + threadIdx.x;
    if (i >= k * c) return;
    int kk = i / c, cc = i % c;
    wt[cc * k + kk] = f32_to_bf16(w[i]);
}

// ---------------- WMMA GEMM ----------------
// y[m, c] = relu?( hb[m,:] @ Ws + mb[m,:] @ Wn + b[c] )
// Ws/Wn given transposed bf16 [C][128]. One wave per 16x16 tile.
template <int C, bool RELU>
__global__ void k_sage_gemm(const unsigned short* __restrict__ hb,
                            const unsigned short* __restrict__ mb,
                            const unsigned short* __restrict__ wst,
                            const unsigned short* __restrict__ wnt,
                            const float* __restrict__ bias,
                            float* __restrict__ y, int mtiles) {
    constexpr int K  = 128;
    constexpr int NT = C / 16;
    int wave = (blockIdx.x * blockDim.x + threadIdx.x) >> 5;
    if (wave >= mtiles * NT) return;           // wave-uniform guard: EXEC stays all-1s
    int lane = threadIdx.x & 31;
    int mt = wave / NT;
    int nt = wave - mt * NT;

    int m0   = mt * 16;
    int col  = nt * 16 + (lane & 15);
    int rowA = m0 + (lane & 15);
    int hiA  = (lane >> 4) ? 8  : 0;   // A-matrix K split (ISA 16-bit A 16x32 layout)
    int hiB  = (lane >> 4) ? 8  : 0;   // B-matrix K split (same as A layout)

    float bcol = bias[col];
    v8f acc = {bcol, bcol, bcol, bcol, bcol, bcol, bcol, bcol};

    const unsigned short* hrow = hb  + (size_t)rowA * K;
    const unsigned short* mrow = mb  + (size_t)rowA * K;
    const unsigned short* wsc  = wst + (size_t)col  * K;
    const unsigned short* wnc  = wnt + (size_t)col  * K;

#pragma unroll
    for (int k0 = 0; k0 < K; k0 += 32) {
        ABOp aS, aM, bS, bN;
        int klo = k0 + hiA;
        aS.q[0] = *(const uint4*)(hrow + klo);
        aS.q[1] = *(const uint4*)(hrow + klo + 16);
        aM.q[0] = *(const uint4*)(mrow + klo);
        aM.q[1] = *(const uint4*)(mrow + klo + 16);
        int kb = k0 + hiB;
        bS.q[0] = *(const uint4*)(wsc + kb);
        bS.q[1] = *(const uint4*)(wsc + kb + 16);
        bN.q[0] = *(const uint4*)(wnc + kb);
        bN.q[1] = *(const uint4*)(wnc + kb + 16);
        acc = __builtin_amdgcn_wmma_f32_16x16x32_bf16(false, aS.v, false, bS.v,
                                                      (short)0, acc, false, false);
        acc = __builtin_amdgcn_wmma_f32_16x16x32_bf16(false, aM.v, false, bN.v,
                                                      (short)0, acc, false, false);
    }

    int rbase = m0 + ((lane >> 4) ? 8 : 0);    // D-matrix: VGPR r -> M = r (+8 hi lanes)
#pragma unroll
    for (int r = 0; r < 8; ++r) {
        float v = acc[r];
        if (RELU) v = fmaxf(v, 0.0f);
        y[(size_t)(rbase + r) * C + col] = v;
    }
}

inline int cdiv(int a, int b) { return (a + b - 1) / b; }

} // namespace

extern "C" void kernel_launch(void* const* d_in, const int* in_sizes, int n_in,
                              void* d_out, int out_size, void* d_ws, size_t ws_size,
                              hipStream_t stream) {
    (void)in_sizes; (void)n_in; (void)out_size; (void)ws_size;

    const float* feat = (const float*)d_in[0];
    const int*   src  = (const int*)d_in[1];
    const int*   dst  = (const int*)d_in[2];
    const float* Ws0  = (const float*)d_in[3];
    const float* Wn0  = (const float*)d_in[4];
    const float* b0   = (const float*)d_in[5];
    const float* Ws1  = (const float*)d_in[6];
    const float* Wn1  = (const float*)d_in[7];
    const float* b1   = (const float*)d_in[8];
    const float* Ws2  = (const float*)d_in[9];
    const float* Wn2  = (const float*)d_in[10];
    const float* b2   = (const float*)d_in[11];
    float* out = (float*)d_out;

    // ---- workspace carve (256B aligned) ----
    char* w = (char*)d_ws;
    size_t off = 0;
    auto carve = [&](size_t bytes) -> char* {
        char* p = w + off;
        off = (off + bytes + 255) & ~(size_t)255;
        return p;
    };
    float*          bufA   = (float*)carve((size_t)NNODES * KFEAT * 4);   // 25.6 MB
    float*          bufB   = (float*)carve((size_t)NNODES * KFEAT * 4);   // 25.6 MB
    unsigned short* hb     = (unsigned short*)carve((size_t)NNODES * KFEAT * 2);
    unsigned short* mb     = (unsigned short*)carve((size_t)NNODES * KFEAT * 2);
    float*          invdeg = (float*)carve((size_t)NNODES * 4);
    unsigned short* wsb    = (unsigned short*)carve((size_t)128 * 128 * 2);
    unsigned short* wnb    = (unsigned short*)carve((size_t)128 * 128 * 2);

    const int TPB = 256;
    const int nElem  = NNODES * KFEAT;
    const int mtiles = NNODES / 16;   // 3125 exactly

    // ---- degrees (layer-invariant) ----
    k_zero_f32<<<cdiv(NNODES, TPB), TPB, 0, stream>>>(invdeg, NNODES);
    k_deg_count<<<cdiv(NEDGES, TPB), TPB, 0, stream>>>(dst, invdeg, NEDGES);
    k_invdeg<<<cdiv(NNODES, TPB), TPB, 0, stream>>>(invdeg, NNODES);

    auto layer = [&](const float* x, float* agg, const float* Ws, const float* Wn,
                     const float* bias, float* y, int C, bool relu) {
        k_zero_f32<<<cdiv(nElem, TPB), TPB, 0, stream>>>(agg, nElem);
        k_scatter_add<<<cdiv(NEDGES * 32, TPB), TPB, 0, stream>>>(x, src, dst, agg, NEDGES);
        k_conv_bf16<<<cdiv(nElem, TPB), TPB, 0, stream>>>(x, hb, nElem);
        k_conv_mean_bf16<<<cdiv(nElem, TPB), TPB, 0, stream>>>(agg, invdeg, mb, nElem);
        k_conv_wt<<<cdiv(128 * C, TPB), TPB, 0, stream>>>(Ws, wsb, 128, C);
        k_conv_wt<<<cdiv(128 * C, TPB), TPB, 0, stream>>>(Wn, wnb, 128, C);
        int waves  = mtiles * (C / 16);
        int blocks = cdiv(waves * 32, TPB);
        if (C == 128 && relu)
            k_sage_gemm<128, true><<<blocks, TPB, 0, stream>>>(hb, mb, wsb, wnb, bias, y, mtiles);
        else
            k_sage_gemm<64, false><<<blocks, TPB, 0, stream>>>(hb, mb, wsb, wnb, bias, y, mtiles);
    };

    // layer 0: x = features, agg in bufA, y0 -> bufA (agg dead after bf16 convert)
    layer(feat, bufA, Ws0, Wn0, b0, bufA, 128, true);
    // layer 1: x = bufA, agg in bufB, y1 -> bufB
    layer(bufA, bufB, Ws1, Wn1, b1, bufB, 128, true);
    // layer 2: x = bufB, agg in bufA, y2 -> d_out (N x 64, no relu)
    layer(bufB, bufA, Ws2, Wn2, b2, out, 64, false);
}
